// QuantumKANClassifier_25220047962617
// MI455X (gfx1250) — matrix operand order repaired and checked by reading, not compiled, not run on hardware
//
#include <hip/hip_runtime.h>
#include <hip/hip_bf16.h>

// ---------------------------------------------------------------------------
// QuantumKANClassifier for MI455X (gfx1250, wave32, WMMA)
//
//   theta = arccos(clip(X));  qsvt = cos(k*theta)  ==  T_k(clip(X))  (Chebyshev)
//   features[b,f] = sum_d T_{d+1}(x) * w[f,d] / (sum_d |w[f,d]| + eps)
//   t = tanh(features);  basis = T_j(clip(t)), j=0..5
//   logits[b,o] = sum_{f,j} basis[b,f,j] * kan_coeffs[o,f,j]
//
// The final contraction is a (16384 x 3072) * (3072 x 2) GEMM -> done with
// v_wmma_f32_16x16x32_f16 (K = feature*8 + j, j padded to 8; N padded to 16).
// All basis math is f32 VALU (Chebyshev recurrences instead of arccos/cos);
// tanh uses the gfx1250 native V_TANH_F32 (or branchless exp2+rcp fallback).
// ---------------------------------------------------------------------------

#define B_ROWS 16384
#define NFEAT  512
#define NDEG   5
#define KEPS   1e-6f

typedef __attribute__((ext_vector_type(16))) _Float16 v16h;
typedef __attribute__((ext_vector_type(8)))  _Float16 v8h;
typedef __attribute__((ext_vector_type(8)))  float    v8f;

__device__ __forceinline__ float fast_tanh(float x) {
#if __has_builtin(__builtin_amdgcn_tanhf)
  return __builtin_amdgcn_tanhf(x);          // native v_tanh_f32 on gfx1250
#elif __has_builtin(__builtin_amdgcn_tanh_f32)
  return __builtin_amdgcn_tanh_f32(x);
#else
  // branchless: tanh(x) = (e^{2x}-1)/(e^{2x}+1), 2*log2(e) = 2.885390081777927
  const float e2 = __builtin_amdgcn_exp2f(x * 2.8853900817779268f);
  return (e2 - 1.0f) * __builtin_amdgcn_rcpf(e2 + 1.0f);
#endif
}

// ---------------------------------------------------------------------------
// Prep: fold 1/norm into the LCU weights (padded to 8 floats per feature) and
// pack kan_coeffs into f16 [o][f][8] (j=0..5 real, j=6,7 zero) for direct
// 16-byte B-fragment loads.
// ---------------------------------------------------------------------------
__global__ __launch_bounds__(256) void qkan_prep(const float* __restrict__ lcu,
                                                 const float* __restrict__ kanc,
                                                 float* __restrict__ wn,
                                                 _Float16* __restrict__ cpk) {
  const int f = blockIdx.x * blockDim.x + threadIdx.x;
  if (f >= NFEAT) return;
  float w[NDEG];
  float s = KEPS;
  #pragma unroll
  for (int d = 0; d < NDEG; ++d) {
    w[d] = lcu[f * NDEG + d];
    s += fabsf(w[d]);
  }
  const float inv = 1.0f / s;
  #pragma unroll
  for (int d = 0; d < NDEG; ++d) wn[f * 8 + d] = w[d] * inv;
  #pragma unroll
  for (int d = NDEG; d < 8; ++d) wn[f * 8 + d] = 0.0f;
  #pragma unroll
  for (int o = 0; o < 2; ++o) {
    #pragma unroll
    for (int j = 0; j < 6; ++j)
      cpk[(o * NFEAT + f) * 8 + j] = (_Float16)kanc[(o * NFEAT + f) * 6 + j];
    cpk[(o * NFEAT + f) * 8 + 6] = (_Float16)0.0f;
    cpk[(o * NFEAT + f) * 8 + 7] = (_Float16)0.0f;
  }
}

// ---------------------------------------------------------------------------
// Per-(row, feature) Chebyshev pipeline: T_1..T_5(x) dot scaled weights,
// tanh, then T_0..T_5(t). cos(k*arccos(x)) == T_k(x) exactly on [-1,1].
// ---------------------------------------------------------------------------
__device__ __forceinline__ void cheb_basis6(const float* __restrict__ X,
                                            const float* __restrict__ wn,
                                            int row, int f, float* out) {
  float x = X[row * NFEAT + f];
  x = fminf(fmaxf(x, -1.0f + KEPS), 1.0f - KEPS);
  const float t1 = x;
  const float t2 = __builtin_fmaf(2.0f * x, t1, -1.0f);
  const float t3 = __builtin_fmaf(2.0f * x, t2, -t1);
  const float t4 = __builtin_fmaf(2.0f * x, t3, -t2);
  const float t5 = __builtin_fmaf(2.0f * x, t4, -t3);
  const float4 w0 = *(const float4*)(wn + f * 8);
  const float  w4 = wn[f * 8 + 4];
  const float feat = t1 * w0.x + t2 * w0.y + t3 * w0.z + t4 * w0.w + t5 * w4;
  float t = fast_tanh(feat);
  t = fminf(fmaxf(t, -1.0f + KEPS), 1.0f - KEPS);
  out[0] = 1.0f;
  out[1] = t;
  out[2] = __builtin_fmaf(2.0f * t, t,      -1.0f);
  out[3] = __builtin_fmaf(2.0f * t, out[2], -t);
  out[4] = __builtin_fmaf(2.0f * t, out[3], -out[2]);
  out[5] = __builtin_fmaf(2.0f * t, out[4], -out[3]);
}

// ---------------------------------------------------------------------------
// Main: 256 threads = 8 waves = 2 row-tiles (16 rows each) x 4 K-slices.
// Each wave runs 32 WMMAs (128 features x 8 j-slots / K=32), then the 4
// slices of a tile are summed through LDS. Grid = 16384/32 = 512 blocks.
// ---------------------------------------------------------------------------
__global__ __launch_bounds__(256) void qkan_main(const float* __restrict__ X,
                                                 const float* __restrict__ wn,
                                                 const _Float16* __restrict__ cpk,
                                                 float* __restrict__ out) {
  __shared__ float red[8][32][8];

  const int lane  = threadIdx.x & 31;
  const int wave  = threadIdx.x >> 5;
  const int slice = wave & 3;                   // K split over features
  const int tile  = blockIdx.x * 2 + (wave >> 2);
  const int b0    = tile * 16;
  const int row   = b0 + (lane & 15);           // A-matrix row M
  const int n     = lane & 15;                  // D column owned by this lane
  const int hi    = lane >> 4;                  // half-wave selector

  v8f acc = {};
  const v8h* cp = (const v8h*)cpk;

  const int f_begin = slice * (NFEAT / 4);      // 128 features per slice
  #pragma unroll 2
  for (int c = 0; c < (NFEAT / 4) / 4; ++c) {   // 4 features per WMMA
    const int fb = f_begin + c * 4;

    // ---- A fragment (16x32 f16): lanes<16 own features fb+0, fb+2 (K 0..7,
    //      16..23); lanes>=16 own fb+1, fb+3 (K 8..15, 24..31).
    const int fA0 = fb + hi;
    float ba[6], bb[6];
    cheb_basis6(X, wn, row, fA0,     ba);
    cheb_basis6(X, wn, row, fA0 + 2, bb);
    v16h av;
    av[0]  = (_Float16)ba[0]; av[1]  = (_Float16)ba[1]; av[2]  = (_Float16)ba[2];
    av[3]  = (_Float16)ba[3]; av[4]  = (_Float16)ba[4]; av[5]  = (_Float16)ba[5];
    av[6]  = (_Float16)0.0f;  av[7]  = (_Float16)0.0f;
    av[8]  = (_Float16)bb[0]; av[9]  = (_Float16)bb[1]; av[10] = (_Float16)bb[2];
    av[11] = (_Float16)bb[3]; av[12] = (_Float16)bb[4]; av[13] = (_Float16)bb[5];
    av[14] = (_Float16)0.0f;  av[15] = (_Float16)0.0f;

    // ---- B fragment (32x16 f16): column n = output o (only n<2 nonzero);
    //      lanes<16 hold K 0..7,16..23 (features fb+0, fb+2), lanes>=16 K 8..15,24..31.
    v16h bv = {};
    if (n < 2) {
      const int fB0 = fb + hi;
      union { v16h v; v8h h[2]; } u;
      u.h[0] = cp[n * NFEAT + fB0];
      u.h[1] = cp[n * NFEAT + fB0 + 2];
      bv = u.v;
    }

    acc = __builtin_amdgcn_wmma_f32_16x16x32_f16(
        /*neg_a=*/false, av, /*neg_b=*/false, bv,
        /*c_mod=*/(short)0, acc, /*reuse_a=*/false, /*reuse_b=*/false);
  }

  // ---- combine the 4 K-slices of each tile through LDS (deterministic)
  #pragma unroll
  for (int r = 0; r < 8; ++r) red[wave][lane][r] = acc[r];
  __syncthreads();

  if (slice == 0 && n < 2) {
    #pragma unroll
    for (int r = 0; r < 8; ++r) {
      const float s = red[wave][lane][r] + red[wave + 1][lane][r] +
                      red[wave + 2][lane][r] + red[wave + 3][lane][r];
      const int m = b0 + hi * 8 + r;            // D rows: lanes>=16 hold M=8..15
      out[m * 2 + n] = s;
    }
  }
}

extern "C" void kernel_launch(void* const* d_in, const int* in_sizes, int n_in,
                              void* d_out, int out_size, void* d_ws, size_t ws_size,
                              hipStream_t stream) {
  (void)in_sizes; (void)n_in; (void)out_size; (void)ws_size;
  const float* X    = (const float*)d_in[0];   // (16384, 512)
  const float* lcu  = (const float*)d_in[1];   // (512, 5)
  const float* kanc = (const float*)d_in[2];   // (2, 512, 6)
  float* out = (float*)d_out;                  // (16384, 2)

  float*    wn  = (float*)d_ws;                        // 512*8 f32  = 16 KB
  _Float16* cpk = (_Float16*)((char*)d_ws + 16 * 1024); // 2*512*8 f16 = 16 KB

  qkan_prep<<<2, 256, 0, stream>>>(lcu, kanc, wn, cpk);
  qkan_main<<<B_ROWS / 32, 256, 0, stream>>>(X, wn, (const _Float16*)cpk, out);
}
